// EncoderLayer_52733608460923
// MI455X (gfx1250) — compile-verified, matrix operand order checked by reading, not run on hardware
//
#include <hip/hip_runtime.h>

typedef __attribute__((ext_vector_type(16))) _Float16 v16h;
typedef __attribute__((ext_vector_type(8)))  _Float16 v8h;
typedef __attribute__((ext_vector_type(8)))  float    v8f;

// ---------------------------------------------------------------------------
// Async global -> LDS staging (CDNA5).  VDST VGPR carries the LDS byte
// address (generic-pointer low 32 bits == wave LDS offset per aperture rules),
// address pair carries the 64-bit global address.  Tracked by ASYNCcnt.
// ---------------------------------------------------------------------------
__device__ __forceinline__ void async_ld_b128(const void* lds, const void* g) {
  unsigned dst = (unsigned)(uintptr_t)lds;
  unsigned long long src = (unsigned long long)(uintptr_t)g;
  asm volatile("global_load_async_to_lds_b128 %0, %1, off"
               :: "v"(dst), "v"(src) : "memory");
}
__device__ __forceinline__ void wait_async0() {
  asm volatile("s_wait_asynccnt 0x0" ::: "memory");
}

// ---------------------------------------------------------------------------
// Fragment loader: 16x32 f16 A-style / Bt-style operand from row-major
// [rows][ld] half buffer.  CDNA5 16-bit A layout (ISA 7.12.2):
//   lanes 0-15 : row = lane,    K = {koff+0..7, koff+16..23}
//   lanes 16-31: row = lane-16, K = {koff+8..15, koff+24..31}
// ---------------------------------------------------------------------------
__device__ __forceinline__ v16h frag_ld(const _Float16* base, int row, int ld,
                                        int koff, int lane) {
  int r  = row + (lane & 15);
  int kb = koff + ((lane & 16) ? 8 : 0);
  const _Float16* p = base + (size_t)r * ld + kb;
  v8h lo = *(const v8h*)(p);
  v8h hi = *(const v8h*)(p + 16);
  v16h out;
#pragma unroll
  for (int i = 0; i < 8; ++i) { out[i] = lo[i]; out[i + 8] = hi[i]; }
  return out;
}

// ---------------------------------------------------------------------------
// f32 -> f16 elementwise
// ---------------------------------------------------------------------------
__global__ void cvt_f32_f16(const float* __restrict__ in,
                            _Float16* __restrict__ out, size_t n) {
  size_t i = (size_t)blockIdx.x * blockDim.x + threadIdx.x;
  if (i < n) out[i] = (_Float16)in[i];
}

// ---------------------------------------------------------------------------
// f32 [K][N] -> f16 transposed [N][K]  (64x64 tiles via LDS)
// ---------------------------------------------------------------------------
__global__ __launch_bounds__(256) void cvt_transpose(
    const float* __restrict__ W, _Float16* __restrict__ Wt, int K, int N) {
  __shared__ __align__(16) _Float16 t[64][72];
  const int k0 = blockIdx.y * 64, n0 = blockIdx.x * 64;
#pragma unroll
  for (int i = 0; i < 16; ++i) {
    int u = threadIdx.x + i * 256;        // 0..4095
    int r = u >> 6, c = u & 63;           // r: k, c: n   (coalesced over n)
    t[c][r] = (_Float16)W[(size_t)(k0 + r) * N + n0 + c];
  }
  __syncthreads();
#pragma unroll
  for (int i = 0; i < 16; ++i) {
    int u = threadIdx.x + i * 256;
    int r = u >> 6, c = u & 63;           // r: n, c: k   (coalesced over k)
    Wt[(size_t)(n0 + r) * K + k0 + c] = t[r][c];
  }
}

// ---------------------------------------------------------------------------
// Per-head V transpose:  Vt[(b*H+h)*64 + d][s] = V[b][s][h*64+d]   (f16)
// ---------------------------------------------------------------------------
__global__ __launch_bounds__(256) void transpose_v(
    const _Float16* __restrict__ V, _Float16* __restrict__ Vt,
    int H, int S, int D) {
  __shared__ __align__(16) _Float16 t[64][72];
  const int bh = blockIdx.y, s0 = blockIdx.x * 64;
  const int b = bh / H, h = bh % H;
  const _Float16* src = V + (size_t)b * S * D + (size_t)h * 64;
#pragma unroll
  for (int i = 0; i < 2; ++i) {
    int u = threadIdx.x + i * 256;        // 0..511
    int r = u >> 3, dc = (u & 7) << 3;    // r: s row, dc: d chunk
    v8h x = *(const v8h*)&src[(size_t)(s0 + r) * D + dc];
#pragma unroll
    for (int j = 0; j < 8; ++j) t[dc + j][r] = x[j];
  }
  __syncthreads();
  _Float16* dst = Vt + (size_t)bh * 64 * S;
#pragma unroll
  for (int i = 0; i < 2; ++i) {
    int u = threadIdx.x + i * 256;
    int r = u >> 3, sc = (u & 7) << 3;    // r: d row, sc: s chunk
    *(v8h*)&dst[(size_t)r * S + s0 + sc] = *(const v8h*)&t[r][sc];
  }
}

// ---------------------------------------------------------------------------
// Tiled WMMA GEMM:  C[M,N] = alpha * (A[M,K] * Bt[N,K]^T + bias)
// Block tile 128x256, K-step 32, 256 threads = 8 waves (2x4), wave tile 64x64.
// Double-buffered LDS filled by async global->LDS DMA.
// SCALEQ folds the attention 1/sqrt(hd) into the Q-projection epilogue.
// ---------------------------------------------------------------------------
#define GBM 128
#define GBN 256
#define GBK 32
#define GLD 40

template <bool RELU, bool OUT16, bool SCALEQ>
__global__ __launch_bounds__(256) void gemm_f16(
    const _Float16* __restrict__ A,   // [M][K]
    const _Float16* __restrict__ Bt,  // [N][K]
    const float* __restrict__ bias, float* __restrict__ Cf,
    _Float16* __restrict__ Ch, int M, int N, int K) {
  __shared__ __align__(16) _Float16 As[2][GBM][GLD];
  __shared__ __align__(16) _Float16 Bs[2][GBN][GLD];

  const int tid  = threadIdx.x;
  const int lane = tid & 31;
  const int wave = tid >> 5;
  const int wm = wave >> 2, wn = wave & 3;     // 2 x 4 waves, 64x64 each
  const int bm0 = blockIdx.y * GBM;
  const int bn0 = blockIdx.x * GBN;

  auto stage = [&](int k0, int buf) {
#pragma unroll
    for (int i = 0; i < 2; ++i) {              // A: 128x32 halves
      int u = tid + i * 256;
      int r = u >> 2, kc = (u & 3) << 3;
      async_ld_b128(&As[buf][r][kc], &A[(size_t)(bm0 + r) * K + k0 + kc]);
    }
#pragma unroll
    for (int i = 0; i < 4; ++i) {              // Bt: 256x32 halves
      int u = tid + i * 256;
      int r = u >> 2, kc = (u & 3) << 3;
      async_ld_b128(&Bs[buf][r][kc], &Bt[(size_t)(bn0 + r) * K + k0 + kc]);
    }
  };

  v8f acc[4][4];
#pragma unroll
  for (int mi = 0; mi < 4; ++mi)
#pragma unroll
    for (int ni = 0; ni < 4; ++ni) acc[mi][ni] = {};

  stage(0, 0);
  wait_async0();
  __syncthreads();

  const int nk = K / GBK;
  int cur = 0;
  for (int k = 0; k < nk; ++k) {
    if (k + 1 < nk) stage((k + 1) * GBK, cur ^ 1);

    v16h bf[4];
#pragma unroll
    for (int ni = 0; ni < 4; ++ni)
      bf[ni] = frag_ld(&Bs[cur][0][0], wn * 64 + ni * 16, GLD, 0, lane);
#pragma unroll
    for (int mi = 0; mi < 4; ++mi) {
      v16h af = frag_ld(&As[cur][0][0], wm * 64 + mi * 16, GLD, 0, lane);
#pragma unroll
      for (int ni = 0; ni < 4; ++ni)
        acc[mi][ni] = __builtin_amdgcn_wmma_f32_16x16x32_f16(
            false, af, false, bf[ni], (short)0, acc[mi][ni], false, false);
    }
    wait_async0();
    __syncthreads();
    cur ^= 1;
  }

  const int colLane = lane & 15;
  const int rowHalf = (lane & 16) ? 8 : 0;
#pragma unroll
  for (int ni = 0; ni < 4; ++ni) {
    int gn = bn0 + wn * 64 + ni * 16 + colLane;
    float bv = bias ? bias[gn] : 0.f;
#pragma unroll
    for (int mi = 0; mi < 4; ++mi) {
#pragma unroll
      for (int r = 0; r < 8; ++r) {
        int gm = bm0 + wm * 64 + mi * 16 + r + rowHalf;
        float v = acc[mi][ni][r] + bv;
        if (SCALEQ) v *= 0.125f;             // fold 1/sqrt(hd) into Q
        if (RELU)   v = fmaxf(v, 0.f);
        if (OUT16) Ch[(size_t)gm * N + gn] = (_Float16)v;
        else       Cf[(size_t)gm * N + gn] = v;
      }
    }
  }
}

// ---------------------------------------------------------------------------
// Flash-style attention.  Q (pre-scaled by 1/sqrt(hd)), K f16 [B,S,D]
// (head h = cols h*64..h*64+63), Vt f16 [B*H*64][S].
// Grid: (S/128, B*H); 8 waves; each wave owns 16 query rows; 64-key K/Vt
// chunks double-buffered in LDS via async DMA.
// ---------------------------------------------------------------------------
#define ACH 64
#define ALD 72
__global__ __launch_bounds__(256) void flash_attn(
    const _Float16* __restrict__ Q, const _Float16* __restrict__ Km,
    const _Float16* __restrict__ Vt, const int* __restrict__ mask,
    _Float16* __restrict__ O, int H, int S, int D) {
  __shared__ __align__(16) _Float16 Ks[2][ACH][ALD];   // [key][d]
  __shared__ __align__(16) _Float16 Vs[2][64][ALD];    // [d][key]
  __shared__ __align__(16) _Float16 Ps[8][16][ALD];    // wave-private P

  const int tid  = threadIdx.x;
  const int lane = tid & 31;
  const int wave = tid >> 5;
  const int colLane = lane & 15;
  const int rowHalf = (lane & 16) ? 8 : 0;

  const int bh = blockIdx.y;
  const int b = bh / H, h = bh % H;
  const int hd = 64;
  const int q0 = blockIdx.x * 128 + wave * 16;

  const _Float16* Qb = Q  + (size_t)b * S * D + (size_t)h * hd;
  const _Float16* Kb = Km + (size_t)b * S * D + (size_t)h * hd;
  const _Float16* Vb = Vt + (size_t)bh * hd * S;       // row d, contiguous in s

  auto stage = [&](int c0, int buf) {
#pragma unroll
    for (int i = 0; i < 2; ++i) {            // K: 64 keys x 64 halves
      int u = tid + i * 256;
      int r = u >> 3, kc = (u & 7) << 3;
      async_ld_b128(&Ks[buf][r][kc], &Kb[(size_t)(c0 + r) * D + kc]);
    }
#pragma unroll
    for (int i = 0; i < 2; ++i) {            // Vt: 64 d-rows x 64 keys
      int u = tid + i * 256;
      int r = u >> 3, sc = (u & 7) << 3;
      async_ld_b128(&Vs[buf][r][sc], &Vb[(size_t)r * S + c0 + sc]);
    }
  };

  // ---- Q fragments straight from global (2 k-steps over hd=64) ----
  v16h aq[2];
  {
    int r  = q0 + colLane;
    int kb = (lane & 16) ? 8 : 0;
    const _Float16* qp = Qb + (size_t)r * D;
#pragma unroll
    for (int ks = 0; ks < 2; ++ks) {
      v8h lo = *(const v8h*)(qp + ks * 32 + kb);
      v8h hi = *(const v8h*)(qp + ks * 32 + kb + 16);
      v16h f;
#pragma unroll
      for (int i = 0; i < 8; ++i) { f[i] = lo[i]; f[i + 8] = hi[i]; }
      aq[ks] = f;
    }
  }

  v8f o[4];
#pragma unroll
  for (int oi = 0; oi < 4; ++oi) o[oi] = {};
  float mrun[8], lrun[8];
#pragma unroll
  for (int r = 0; r < 8; ++r) { mrun[r] = -1e30f; lrun[r] = 0.f; }

  stage(0, 0);
  wait_async0();
  __syncthreads();

  int cur = 0;
  for (int c0 = 0; c0 < S; c0 += ACH) {
    if (c0 + ACH < S) stage(c0 + ACH, cur ^ 1);

    // ---- scores = Qhat * K^T  (all 4 B-fragments preloaded per k-step) ----
    v8f s[4];
#pragma unroll
    for (int ni = 0; ni < 4; ++ni) s[ni] = {};
#pragma unroll
    for (int ks = 0; ks < 2; ++ks) {
      v16h bf[4];
#pragma unroll
      for (int ni = 0; ni < 4; ++ni)
        bf[ni] = frag_ld(&Ks[cur][0][0], ni * 16, ALD, ks * 32, lane);
#pragma unroll
      for (int ni = 0; ni < 4; ++ni)
        s[ni] = __builtin_amdgcn_wmma_f32_16x16x32_f16(
            false, aq[ks], false, bf[ni], (short)0, s[ni], false, false);
    }

    // ---- mask (scale already folded into Q) ----
#pragma unroll
    for (int ni = 0; ni < 4; ++ni) {
      int gk = c0 + ni * 16 + colLane;
      bool ok = mask[(size_t)b * S + gk] != 0;
#pragma unroll
      for (int r = 0; r < 8; ++r) s[ni][r] = ok ? s[ni][r] : -1e9f;
    }

    // ---- online softmax (row stats per 16-lane half-wave) ----
#pragma unroll
    for (int r = 0; r < 8; ++r) {
      float mx = fmaxf(fmaxf(s[0][r], s[1][r]), fmaxf(s[2][r], s[3][r]));
#pragma unroll
      for (int off = 1; off < 16; off <<= 1)
        mx = fmaxf(mx, __shfl_xor(mx, off, 32));
      float nm = fmaxf(mrun[r], mx);
      float sum = 0.f;
#pragma unroll
      for (int ni = 0; ni < 4; ++ni) {
        float e = __expf(s[ni][r] - nm);
        s[ni][r] = e;
        sum += e;
      }
#pragma unroll
      for (int off = 1; off < 16; off <<= 1)
        sum += __shfl_xor(sum, off, 32);
      float alpha = __expf(mrun[r] - nm);
      lrun[r] = lrun[r] * alpha + sum;
      mrun[r] = nm;
#pragma unroll
      for (int oi = 0; oi < 4; ++oi) o[oi][r] *= alpha;
    }

    // ---- P: C-layout -> A-layout via wave-private LDS ----
    _Float16(*myP)[ALD] = Ps[wave];
#pragma unroll
    for (int ni = 0; ni < 4; ++ni)
#pragma unroll
      for (int r = 0; r < 8; ++r)
        myP[r + rowHalf][ni * 16 + colLane] = (_Float16)s[ni][r];

    // ---- O += P * V  (A + all 4 B fragments preloaded per k-step) ----
#pragma unroll
    for (int ks = 0; ks < 2; ++ks) {
      v16h ap = frag_ld(&myP[0][0], 0, ALD, ks * 32, lane);
      v16h bv[4];
#pragma unroll
      for (int oi = 0; oi < 4; ++oi)
        bv[oi] = frag_ld(&Vs[cur][0][0], oi * 16, ALD, ks * 32, lane);
#pragma unroll
      for (int oi = 0; oi < 4; ++oi)
        o[oi] = __builtin_amdgcn_wmma_f32_16x16x32_f16(
            false, ap, false, bv[oi], (short)0, o[oi], false, false);
    }
    wait_async0();
    __syncthreads();
    cur ^= 1;
  }

  // ---- normalize, emit f16 [B,S,D] ----
#pragma unroll
  for (int oi = 0; oi < 4; ++oi) {
    int gd = h * hd + oi * 16 + colLane;
#pragma unroll
    for (int r = 0; r < 8; ++r) {
      int gq = q0 + r + rowHalf;
      float v = o[oi][r] / lrun[r];
      O[((size_t)b * S + gq) * D + gd] = (_Float16)v;
    }
  }
}

// ---------------------------------------------------------------------------
// Residual add + LayerNorm, one block per token row (values cached in regs).
// ---------------------------------------------------------------------------
__global__ __launch_bounds__(256) void residual_ln(
    const float* __restrict__ X, const float* __restrict__ Y,
    const float* __restrict__ g, const float* __restrict__ be,
    float* __restrict__ outF, _Float16* __restrict__ outH, int D) {
  __shared__ float sbuf[8], ssbuf[8];
  const int tid = threadIdx.x, lane = tid & 31, wave = tid >> 5;
  const size_t row = blockIdx.x;
  const float* xr = X + row * D;
  const float* yr = Y + row * D;

  float vals[8];
  float s = 0.f, ss = 0.f;
  int cnt = 0;
  for (int i = tid; i < D; i += 256) {
    float v = xr[i] + yr[i];
    vals[cnt++] = v;
    s += v; ss += v * v;
  }
#pragma unroll
  for (int off = 16; off > 0; off >>= 1) {
    s  += __shfl_xor(s,  off, 32);
    ss += __shfl_xor(ss, off, 32);
  }
  if (lane == 0) { sbuf[wave] = s; ssbuf[wave] = ss; }
  __syncthreads();
  float ts = 0.f, tss = 0.f;
#pragma unroll
  for (int w = 0; w < 8; ++w) { ts += sbuf[w]; tss += ssbuf[w]; }
  float mu  = ts / D;
  float var = tss / D - mu * mu;
  float rin = rsqrtf(var + 1e-5f);

  cnt = 0;
  for (int i = tid; i < D; i += 256) {
    float v = (vals[cnt++] - mu) * rin * g[i] + be[i];
    outF[row * D + i] = v;
    if (outH) outH[row * D + i] = (_Float16)v;
  }
}

// ---------------------------------------------------------------------------
// Host orchestration
// ---------------------------------------------------------------------------
extern "C" void kernel_launch(void* const* d_in, const int* in_sizes, int n_in,
                              void* d_out, int out_size, void* d_ws,
                              size_t ws_size, hipStream_t stream) {
  const float* x   = (const float*)d_in[0];
  const int*   mask= (const int*)  d_in[1];
  const float* Wq  = (const float*)d_in[2];
  const float* bq  = (const float*)d_in[3];
  const float* Wk  = (const float*)d_in[4];
  const float* bk  = (const float*)d_in[5];
  const float* Wv  = (const float*)d_in[6];
  const float* bv  = (const float*)d_in[7];
  const float* Wo  = (const float*)d_in[8];
  const float* bo  = (const float*)d_in[9];
  const float* W1  = (const float*)d_in[10];
  const float* b1  = (const float*)d_in[11];
  const float* W2  = (const float*)d_in[12];
  const float* b2  = (const float*)d_in[13];
  const float* g1  = (const float*)d_in[14];
  const float* be1 = (const float*)d_in[15];
  const float* g2  = (const float*)d_in[16];
  const float* be2 = (const float*)d_in[17];

  constexpr int Bb = 4, S = 2048, D = 1024, H = 16, F = 4096;
  const size_t T = (size_t)Bb * S;   // 8192 tokens

  size_t off = 0;
  auto take = [&](size_t bytes) {
    void* p = (char*)d_ws + off;
    off += (bytes + 255) & ~(size_t)255;
    return p;
  };
  _Float16* x16  = (_Float16*)take(T * D * 2);
  _Float16* Wqt  = (_Float16*)take((size_t)D * D * 2);   // all weights stored
  _Float16* Wkt  = (_Float16*)take((size_t)D * D * 2);   // transposed [N][K]
  _Float16* Wvt  = (_Float16*)take((size_t)D * D * 2);
  _Float16* Wot  = (_Float16*)take((size_t)D * D * 2);
  _Float16* W1t  = (_Float16*)take((size_t)D * F * 2);
  _Float16* W2t  = (_Float16*)take((size_t)F * D * 2);
  size_t qkvOff = off;                 // Q/K/V/O region, later reused for ff1
  _Float16* Q16 = (_Float16*)take(T * D * 2);
  _Float16* K16 = (_Float16*)take(T * D * 2);
  _Float16* V16 = (_Float16*)take(T * D * 2);
  _Float16* O16 = (_Float16*)take(T * D * 2);
  _Float16* Vt  = (_Float16*)take(T * D * 2);            // [B*H*64][S]
  _Float16* ff1 = (_Float16*)((char*)d_ws + qkvOff);     // T*F*2 == 4*(T*D*2)
  float*    tmpF = (float*)take(T * D * 4);              // attn_out, later ff2
  float*    hF   = (float*)take(T * D * 4);
  _Float16* h16  = (_Float16*)take(T * D * 2);
  (void)ws_size; (void)n_in; (void)in_sizes; (void)out_size;

  cvt_f32_f16<<<(unsigned)((T * D + 255) / 256), 256, 0, stream>>>(x, x16,
                                                                   T * D);
  {  // weights: convert + transpose
    dim3 gdd(D / 64, D / 64);
    cvt_transpose<<<gdd, 256, 0, stream>>>(Wq, Wqt, D, D);
    cvt_transpose<<<gdd, 256, 0, stream>>>(Wk, Wkt, D, D);
    cvt_transpose<<<gdd, 256, 0, stream>>>(Wv, Wvt, D, D);
    cvt_transpose<<<gdd, 256, 0, stream>>>(Wo, Wot, D, D);
    cvt_transpose<<<dim3(F / 64, D / 64), 256, 0, stream>>>(W1, W1t, D, F);
    cvt_transpose<<<dim3(D / 64, F / 64), 256, 0, stream>>>(W2, W2t, F, D);
  }

  // -------- QKV projections (Q pre-scaled by 1/sqrt(hd)) --------
  {
    dim3 g(D / GBN, (unsigned)(T / GBM));
    gemm_f16<false, true, true ><<<g, 256, 0, stream>>>(x16, Wqt, bq, nullptr,
                                                        Q16, (int)T, D, D);
    gemm_f16<false, true, false><<<g, 256, 0, stream>>>(x16, Wkt, bk, nullptr,
                                                        K16, (int)T, D, D);
    gemm_f16<false, true, false><<<g, 256, 0, stream>>>(x16, Wvt, bv, nullptr,
                                                        V16, (int)T, D, D);
  }

  // -------- V per-head transpose, then attention --------
  transpose_v<<<dim3(S / 64, Bb * H), 256, 0, stream>>>(V16, Vt, H, S, D);
  flash_attn<<<dim3(S / 128, Bb * H), 256, 0, stream>>>(Q16, K16, Vt, mask,
                                                        O16, H, S, D);

  // -------- output projection -> tmpF --------
  {
    dim3 g(D / GBN, (unsigned)(T / GBM));
    gemm_f16<false, false, false><<<g, 256, 0, stream>>>(O16, Wot, bo, tmpF,
                                                         nullptr, (int)T, D, D);
  }

  // -------- residual + LN1 -> hF (f32) and h16 (f16) --------
  residual_ln<<<(unsigned)T, 256, 0, stream>>>(x, tmpF, g1, be1, hF, h16, D);

  // -------- FFN1 (ReLU) -> ff1 --------
  {
    dim3 g(F / GBN, (unsigned)(T / GBM));
    gemm_f16<true, true, false><<<g, 256, 0, stream>>>(h16, W1t, b1, nullptr,
                                                       ff1, (int)T, F, D);
  }

  // -------- FFN2 -> tmpF --------
  {
    dim3 g(D / GBN, (unsigned)(T / GBM));
    gemm_f16<false, false, false><<<g, 256, 0, stream>>>(ff1, W2t, b2, tmpF,
                                                         nullptr, (int)T, D, F);
  }

  // -------- residual + LN2 -> d_out --------
  residual_ln<<<(unsigned)T, 256, 0, stream>>>(hF, tmpF, g2, be2, (float*)d_out,
                                               nullptr, D);
}
